// GraphNeuralNetwork_43233140801672
// MI455X (gfx1250) — matrix operand order repaired and checked by reading, not compiled, not run on hardware
//
#include <hip/hip_runtime.h>
#include <hip/hip_bf16.h>

// B=64, N=512, D=256, L=3. All activations bf16, accumulate f32 via WMMA.

typedef __attribute__((ext_vector_type(16))) __bf16 v16bf;
typedef __attribute__((ext_vector_type(8)))  float  v8f;
typedef int v4i __attribute__((vector_size(16)));

#ifndef __has_builtin
#define __has_builtin(x) 0
#endif
#if __has_builtin(__builtin_amdgcn_global_load_async_to_lds_b128) && \
    __has_builtin(__builtin_amdgcn_s_wait_asynccnt)
#define HAVE_ASYNC_LDS 1
#else
#define HAVE_ASYNC_LDS 0
#endif

#define AS1 __attribute__((address_space(1)))
#define AS3 __attribute__((address_space(3)))

union FragAB {
  v16bf v;
  uint4 q[2];
  unsigned short u[16];
};

__device__ __forceinline__ unsigned short f2bf(float f) {
  unsigned int u = __float_as_uint(f);
  u += 0x7fffu + ((u >> 16) & 1u);   // round-to-nearest-even
  return (unsigned short)(u >> 16);
}

// ---------------- embed lookup: x[row,:] = bf16(embed[clip(fp[row])][:]) ----
__global__ void k_embed(const int* __restrict__ fp, const float* __restrict__ emb,
                        unsigned short* __restrict__ x) {
  int row = blockIdx.x;                 // 0..32767 (b*512+n)
  int f = fp[row];
  f = min(max(f, 0), 9999);
  int d = threadIdx.x;                  // 0..255
  x[row * 256 + d] = f2bf(emb[f * 256 + d]);
}

// ---------------- tiled WMMA GEMM: Out = A(bf16)[M,256] @ W(f32)[256,256] + bias
// mode 0: Out bf16, bias = bias[col]
// mode 1: Out f32,  bias = gmat[(row>>9)*256 + col]   (per-batch bias)
#define GM 128
#define GN 128
#define GK 32
#define APAD 40   // GK + 8 pad (16-lane frag loads hit distinct banks)

__global__ __launch_bounds__(256) void k_gemm(
    const unsigned short* __restrict__ A,
    const float* __restrict__ W,
    const float* __restrict__ bias,
    const float* __restrict__ gmat,
    void* __restrict__ Out, int mode) {
  __shared__ unsigned short As[GM * APAD];   // 10 KB
  __shared__ unsigned short Bt[GN * APAD];   // 10 KB (transposed: Bt[n][k])
  int t = threadIdx.x;
  int w = t >> 5, l = t & 31;
  int lr = l & 15, lh = l >> 4;
  int m0 = blockIdx.x * GM;
  int n0 = blockIdx.y * GN;

  v8f acc[8] = {};

  for (int kb = 0; kb < 256; kb += GK) {
    // prefetch next K-tiles into L2/L0 while this tile computes
    if (kb + GK < 256) {
      if (t < 128)
        __builtin_prefetch(&A[(size_t)(m0 + t) * 256 + kb + GK], 0, 1);
      __builtin_prefetch(&W[(kb + GK + (t >> 3)) * 256 + n0 + (t & 7) * 16], 0, 1);
    }
#if HAVE_ASYNC_LDS
    // async DMA global->LDS for the bf16 A tile (ASYNCcnt-tracked, no VGPR trip)
    {
      int c0_ = t, c1_ = t + 256;
      int r0_ = c0_ >> 2, k0_ = (c0_ & 3) * 8;
      int r1_ = c1_ >> 2, k1_ = (c1_ & 3) * 8;
      __builtin_amdgcn_global_load_async_to_lds_b128(
          (AS1 v4i*)&A[(size_t)(m0 + r0_) * 256 + kb + k0_],
          (AS3 v4i*)&As[r0_ * APAD + k0_], 0, 0);
      __builtin_amdgcn_global_load_async_to_lds_b128(
          (AS1 v4i*)&A[(size_t)(m0 + r1_) * 256 + kb + k1_],
          (AS3 v4i*)&As[r1_ * APAD + k1_], 0, 0);
    }
#else
    #pragma unroll
    for (int c = 0; c < 2; c++) {
      int chunk = t + c * 256;            // 0..511
      int row = chunk >> 2;               // 0..127
      int ko  = (chunk & 3) * 8;          // 0,8,16,24
      *(uint4*)(&As[row * APAD + ko]) =
          *(const uint4*)(&A[(size_t)(m0 + row) * 256 + kb + ko]);
    }
#endif
    // stage W tile 32x128, f32 -> bf16, transposed into Bt[n][k]
    #pragma unroll
    for (int c = 0; c < 16; c++) {
      int idx = t + c * 256;              // 0..4095
      int k = idx >> 7;                   // 0..31
      int n = idx & 127;
      Bt[n * APAD + k] = f2bf(W[(kb + k) * 256 + n0 + n]);
    }
#if HAVE_ASYNC_LDS
    __builtin_amdgcn_s_wait_asynccnt(0);
#endif
    __syncthreads();

    // preload A-frag + all 8 B-frags, then burst 8 WMMAs back-to-back
    FragAB af, bq[8];
    {
      const unsigned short* ap = &As[(w * 16 + lr) * APAD];
      af.q[0] = *(const uint4*)(ap + lh * 8);
      af.q[1] = *(const uint4*)(ap + 16 + lh * 8);
    }
    #pragma unroll
    for (int nt = 0; nt < 8; nt++) {
      const unsigned short* bp = &Bt[(nt * 16 + lr) * APAD + lh * 8];
      bq[nt].q[0] = *(const uint4*)(bp);
      bq[nt].q[1] = *(const uint4*)(bp + 16);
    }
    #pragma unroll
    for (int nt = 0; nt < 8; nt++)
      acc[nt] = __builtin_amdgcn_wmma_f32_16x16x32_bf16(
          false, af.v, false, bq[nt].v, (short)0, acc[nt], false, false);
    __syncthreads();
  }

  #pragma unroll
  for (int nt = 0; nt < 8; nt++) {
    #pragma unroll
    for (int r = 0; r < 8; r++) {
      int row = m0 + w * 16 + lh * 8 + r;
      int col = n0 + nt * 16 + lr;
      float v = acc[nt][r];
      if (mode == 0) {
        v += bias[col];
        ((unsigned short*)Out)[(size_t)row * 256 + col] = f2bf(v);
      } else {
        v += gmat[(row >> 9) * 256 + col];
        ((float*)Out)[(size_t)row * 256 + col] = v;
      }
    }
  }
}

// ---------------- fused attention per (batch, 64-row block) -----------------
// out = L2normalize(relu(softmax(h h^T) @ h)); softmax denominator cancels
// under the L2 normalize, so only a running row-max is tracked (flash-style).
#define CPAD 72   // 64 + 8 pad

__global__ __launch_bounds__(256) void k_attn(
    const unsigned short* __restrict__ H,    // [B*512, 256] bf16
    unsigned short* __restrict__ Xo) {       // [B*512, 256] bf16
  __shared__ unsigned short hCt[256 * CPAD];      // 36 KB: hCt[e][m'] = h[c0+m'][e]
  __shared__ unsigned short Pls[4 * 16 * CPAD];   // 9 KB:  P per row-group
  __shared__ float ssp[2][64];                    // cross-wave sum-of-squares

  int b  = blockIdx.x;
  int rb = blockIdx.y;              // 0..7, 64-row block
  int t = threadIdx.x;
  int w = t >> 5, l = t & 31;
  int g  = w & 3;                   // row group (16 rows)
  int ch = w >> 2;                  // column half of output D (0:0-127, 1:128-255)
  int lr = l & 15, lh = l >> 4;

  const unsigned short* hB = H + (size_t)b * 512 * 256;
  int row0 = rb * 64 + g * 16;

  // A-fragments of this wave's 16 rows of h, all K=256 (8 k-steps), registers
  FragAB ar[8];
  {
    const unsigned short* rp = hB + (size_t)(row0 + lr) * 256;
    #pragma unroll
    for (int ks = 0; ks < 8; ks++) {
      ar[ks].q[0] = *(const uint4*)(rp + ks * 32 + lh * 8);
      ar[ks].q[1] = *(const uint4*)(rp + ks * 32 + 16 + lh * 8);
    }
  }

  v8f acc[8] = {};
  float rmax[8];
  #pragma unroll
  for (int r = 0; r < 8; r++) rmax[r] = -1e30f;

  for (int cb = 0; cb < 8; cb++) {
    int c0 = cb * 64;
    __syncthreads();                      // prior PV reads of hCt/Pls complete
    // stage hCt[e][mc] = h[c0+mc][e]: vector global reads, transposed LDS writes
    {
      int mc = t >> 2;
      int e0 = (t & 3) * 64;
      const unsigned short* rp = hB + (size_t)(c0 + mc) * 256 + e0;
      #pragma unroll
      for (int eo = 0; eo < 64; eo += 8) {
        FragAB tmp;
        tmp.q[0] = *(const uint4*)(rp + eo);
        #pragma unroll
        for (int j = 0; j < 8; j++)
          hCt[(e0 + eo + j) * CPAD + mc] = tmp.u[j];
      }
    }
    if (cb < 7)   // prefetch next column block of h
      __builtin_prefetch(hB + (size_t)(c0 + 64 + (t >> 2)) * 256 + (t & 3) * 64, 0, 1);
    __syncthreads();

    // S = hR @ hC^T  (B-frag = contiguous rows of h, straight from global)
    v8f s[4];
    #pragma unroll
    for (int st = 0; st < 4; st++) {
      v8f z = {};
      #pragma unroll
      for (int kg = 0; kg < 2; kg++) {
        FragAB bq[4];
        #pragma unroll
        for (int j = 0; j < 4; j++) {
          int ks = kg * 4 + j;
          const unsigned short* bp =
              hB + (size_t)(c0 + st * 16 + lr) * 256 + ks * 32 + lh * 8;
          bq[j].q[0] = *(const uint4*)(bp);
          bq[j].q[1] = *(const uint4*)(bp + 16);
        }
        #pragma unroll
        for (int j = 0; j < 4; j++)
          z = __builtin_amdgcn_wmma_f32_16x16x32_bf16(
              false, ar[kg * 4 + j].v, false, bq[j].v, (short)0, z, false, false);
      }
      s[st] = z;
    }

    // flash-softmax (row-max only; denominator cancels under L2 normalize)
    float nm[8];
    #pragma unroll
    for (int r = 0; r < 8; r++) {
      float m = fmaxf(fmaxf(s[0][r], s[1][r]), fmaxf(s[2][r], s[3][r]));
      m = fmaxf(m, __shfl_xor(m, 1));
      m = fmaxf(m, __shfl_xor(m, 2));
      m = fmaxf(m, __shfl_xor(m, 4));
      m = fmaxf(m, __shfl_xor(m, 8));
      nm[r] = fmaxf(rmax[r], m);
    }
    #pragma unroll
    for (int r = 0; r < 8; r++) {
      float alpha = __expf(rmax[r] - nm[r]);
      rmax[r] = nm[r];
      #pragma unroll
      for (int nt = 0; nt < 8; nt++) acc[nt][r] *= alpha;
      #pragma unroll
      for (int st = 0; st < 4; st++) s[st][r] = __expf(s[st][r] - nm[r]);
    }

    // write P to LDS in A-layout source form (one wave per row-group)
    if (ch == 0) {
      unsigned short* pp = &Pls[g * 16 * CPAD];
      #pragma unroll
      for (int st = 0; st < 4; st++)
        #pragma unroll
        for (int r = 0; r < 8; r++)
          pp[(lh * 8 + r) * CPAD + st * 16 + lr] = f2bf(s[st][r]);
    }
    __syncthreads();

    // acc += P @ hC  (K = 64 -> two 32-deep WMMA steps, 4-wide frag bursts)
    #pragma unroll
    for (int ks = 0; ks < 2; ks++) {
      FragAB af;
      const unsigned short* ap = &Pls[g * 16 * CPAD + lr * CPAD + ks * 32];
      af.q[0] = *(const uint4*)(ap + lh * 8);
      af.q[1] = *(const uint4*)(ap + 16 + lh * 8);
      #pragma unroll
      for (int ng = 0; ng < 2; ng++) {
        FragAB bq[4];
        #pragma unroll
        for (int j = 0; j < 4; j++) {
          int nt = ng * 4 + j;
          const unsigned short* bp =
              &hCt[(ch * 128 + nt * 16 + lr) * CPAD + ks * 32 + lh * 8];
          bq[j].q[0] = *(const uint4*)(bp);
          bq[j].q[1] = *(const uint4*)(bp + 16);
        }
        #pragma unroll
        for (int j = 0; j < 4; j++)
          acc[ng * 4 + j] = __builtin_amdgcn_wmma_f32_16x16x32_bf16(
              false, af.v, false, bq[j].v, (short)0, acc[ng * 4 + j], false, false);
      }
    }
  }

  // relu + row sum-of-squares (reduce across 16 lanes, then across col-halves)
  float ss[8];
  #pragma unroll
  for (int r = 0; r < 8; r++) {
    float s2 = 0.f;
    #pragma unroll
    for (int nt = 0; nt < 8; nt++) {
      float v = fmaxf(acc[nt][r], 0.f);
      acc[nt][r] = v;
      s2 += v * v;
    }
    s2 += __shfl_xor(s2, 1);
    s2 += __shfl_xor(s2, 2);
    s2 += __shfl_xor(s2, 4);
    s2 += __shfl_xor(s2, 8);
    ss[r] = s2;
  }
  __syncthreads();
  if (lr == 0) {
    #pragma unroll
    for (int r = 0; r < 8; r++) ssp[ch][g * 16 + lh * 8 + r] = ss[r];
  }
  __syncthreads();

  #pragma unroll
  for (int r = 0; r < 8; r++) {
    float tot = ssp[0][g * 16 + lh * 8 + r] + ssp[1][g * 16 + lh * 8 + r];
    float scale = 1.0f / fmaxf(sqrtf(tot), 1e-12f);
    int row = row0 + lh * 8 + r;
    unsigned short* xo = Xo + ((size_t)b * 512 + row) * 256;
    #pragma unroll
    for (int nt = 0; nt < 8; nt++)
      xo[ch * 128 + nt * 16 + lr] = f2bf(acc[nt][r] * scale);
  }
}

// ---------------- feat = ft @ Wf + bf  ([64,9]x[9,256]) --------------------
__global__ void k_feat(const float* __restrict__ ft, const float* __restrict__ Wf,
                       const float* __restrict__ bfv, float* __restrict__ feat) {
  int b = blockIdx.x, d = threadIdx.x;
  float s = bfv[d];
  #pragma unroll
  for (int j = 0; j < 9; j++) s += ft[b * 9 + j] * Wf[j * 256 + d];
  feat[b * 256 + d] = s;
}

// ---------------- g = feat @ Wp[256:512,:] + bp ([64,256]x[256,256]) -------
__global__ void k_g(const float* __restrict__ feat, const float* __restrict__ Wp,
                    const float* __restrict__ bpv, float* __restrict__ g) {
  int b = blockIdx.x, e = threadIdx.x;
  float s = bpv[e];
  for (int d = 0; d < 256; d++) s += feat[b * 256 + d] * Wp[(256 + d) * 256 + e];
  g[b * 256 + e] = s;
}

extern "C" void kernel_launch(void* const* d_in, const int* in_sizes, int n_in,
                              void* d_out, int out_size, void* d_ws, size_t ws_size,
                              hipStream_t stream) {
  const int*   fp    = (const int*)d_in[0];
  const float* ft    = (const float*)d_in[4];
  const float* emb   = (const float*)d_in[5];
  const float* Wl    = (const float*)d_in[6];
  const float* bl    = (const float*)d_in[7];
  const float* Wf    = (const float*)d_in[8];
  const float* bfv   = (const float*)d_in[9];
  const float* Wp    = (const float*)d_in[10];
  const float* bpv   = (const float*)d_in[11];

  // workspace: x (bf16, 16 MiB) | feat (64x256 f32) | g (64x256 f32)
  unsigned short* x = (unsigned short*)d_ws;
  float* feat = (float*)((char*)d_ws + (size_t)32768 * 256 * 2);
  float* g    = feat + 64 * 256;
  unsigned short* h = (unsigned short*)d_out;   // d_out doubles as h scratch

  k_embed<<<32768, 256, 0, stream>>>(fp, emb, x);
  for (int l = 0; l < 3; l++) {
    k_gemm<<<dim3(256, 2), 256, 0, stream>>>(x, Wl + l * 256 * 256, bl + l * 256,
                                             nullptr, (void*)h, 0);
    k_attn<<<dim3(64, 8), 256, 0, stream>>>(h, x);
  }
  k_feat<<<64, 256, 0, stream>>>(ft, Wf, bfv, feat);
  k_g<<<64, 256, 0, stream>>>(feat, Wp, bpv, g);
  // out = x @ Wp[0:256,:] + g[batch]   (concat folded into two GEMMs)
  k_gemm<<<dim3(256, 2), 256, 0, stream>>>(x, Wp, nullptr, g, d_out, 1);
}
